// CrossOp_14096082665486
// MI455X (gfx1250) — matrix operand order repaired and checked by reading, not compiled, not run on hardware
//
#include <hip/hip_runtime.h>

typedef _Float16 v16h __attribute__((ext_vector_type(16)));
typedef _Float16 v8h  __attribute__((ext_vector_type(8)));
typedef float    v8f  __attribute__((ext_vector_type(8)));

#define NEG_SLOPE 0.01f

// Problem dims (fixed by reference)
#define DB   4
#define DSY  32
#define DCIN 16
#define DCO  64
#define DH   128
#define DW   128
#define KREAL 144          // 9 taps * 16 ch,  k = tap*16 + ch  (tap-major!)
#define KPAD  160          // 5 WMMA chunks of 32
#define KSTR  168          // LDS row stride in halves: 16B aligned, low bank conflict

// Dynamic LDS layout (bytes)
#define WX_OFF    0
#define WY_OFF    (DCO * KSTR * 2)                 // 21504
#define PATCH_OFF (2 * DCO * KSTR * 2)             // 43008
#define SMEM_BYTES (PATCH_OFF + DW * KSTR * 2)     // 86016

// ---------------------------------------------------------------------------
// im2col into LDS: patch[pix][k], k = tap*16 + ch. Fully unrolled over taps so
// kh/kw/bounds are (nearly) compile-time; loads coalesced in pix across lanes.
// Each thread owns pixel (tid&127) and channels of parity (tid>>7).
// ---------------------------------------------------------------------------
__device__ __forceinline__ void build_patch(const float* __restrict__ img, int h,
                                            int pix, int c2,
                                            _Float16* __restrict__ patch) {
    _Float16* prow = patch + pix * KSTR;
    #pragma unroll
    for (int tap = 0; tap < 9; ++tap) {
        const int kh = tap / 3;            // compile-time
        const int kw = tap - kh * 3;       // compile-time
        const int hh = h + kh - 1;
        const int ww = pix + kw - 1;
        const bool ok = (hh >= 0) && (hh < DH) && (ww >= 0) && (ww < DW);
        const float* src = img + (long)hh * DW + ww;
        #pragma unroll
        for (int ch = 0; ch < DCIN; ch += 2) {
            const int c = ch + c2;
            float v = ok ? src[(long)c * (DH * DW)] : 0.0f;
            prow[tap * 16 + c] = (_Float16)v;
        }
    }
}

// ---------------------------------------------------------------------------
// One conv tile: wave's 16 pixels x 64 Co. All LDS addresses are a per-lane
// base + compile-time immediate -> ds_load_b128 with IOFFSET, no VALU math.
//   A lane layout (16-bit): elems 0..7 -> K = kc*32 + hi*8 + (0..7)
//                           elems 8..15 -> K = kc*32 + 16 + hi*8 + (0..7)
//   B lane layout (16-bit): same as A (elems 0..7 -> kc*32 + hi*8, 8..15 -> kc*32 + 16 + hi*8)
// ---------------------------------------------------------------------------
__device__ __forceinline__ void conv_tile(const _Float16* __restrict__ Wl,
                                          const _Float16* __restrict__ patch,
                                          int aBase, int bBase, v8f acc[4]) {
    #pragma unroll
    for (int kc = 0; kc < 5; ++kc) {
        v8h blo = *(const v8h*)(patch + bBase + kc * 32);
        v8h bhi = *(const v8h*)(patch + bBase + kc * 32 + 16);
        v16h bfrag = __builtin_shufflevector(blo, bhi,
                       0,1,2,3,4,5,6,7,8,9,10,11,12,13,14,15);
        #pragma unroll
        for (int ct = 0; ct < 4; ++ct) {
            v8h alo = *(const v8h*)(Wl + aBase + ct * (16 * KSTR) + kc * 32);
            v8h ahi = *(const v8h*)(Wl + aBase + ct * (16 * KSTR) + kc * 32 + 16);
            v16h afrag = __builtin_shufflevector(alo, ahi,
                           0,1,2,3,4,5,6,7,8,9,10,11,12,13,14,15);
            acc[ct] = __builtin_amdgcn_wmma_f32_16x16x32_f16(
                false, afrag, false, bfrag, (short)0, acc[ct], false, false);
        }
    }
}

// ---------------------------------------------------------------------------
// One block per (h, b); 8 waves cover the 128-wide row, each wave 16 px x 64 Co.
// ---------------------------------------------------------------------------
__global__ void __launch_bounds__(256)
cross_conv_kernel(const float* __restrict__ x,     // [4][1][16][128][128]
                  const float* __restrict__ y,     // [4][32][16][128][128]
                  const float* __restrict__ w,     // [64][32][3][3]
                  const float* __restrict__ bias,  // [64]
                  float* __restrict__ out)         // new_target ++ interaction
{
    extern __shared__ char smem[];
    _Float16* Wx    = (_Float16*)(smem + WX_OFF);
    _Float16* Wy    = (_Float16*)(smem + WY_OFF);
    _Float16* patch = (_Float16*)(smem + PATCH_OFF);

    const int tid  = threadIdx.x;
    const int lane = tid & 31;
    const int wave = tid >> 5;
    const int h    = blockIdx.x;
    const int b    = blockIdx.y;
    const int n    = lane & 15;
    const int hi   = lane >> 4;
    const int pix  = wave * 16 + n;          // this lane's output pixel (column)
    const int bpix = tid & 127;              // patch-build pixel
    const int c2   = tid >> 7;               // patch-build channel parity

    // per-lane LDS fragment bases (half-element units), sy-invariant
    const int aBase = n * KSTR + hi * 8;
    const int bBase = pix * KSTR + hi * 8;

    // ---- stage weights in tap-major K order, zero-padded ----
    for (int idx = tid; idx < DCO * KSTR; idx += 256) {
        const int co = idx / KSTR;
        const int k  = idx - co * KSTR;
        float vx = 0.0f, vy = 0.0f;
        if (k < KREAL) {
            const int tap = k >> 4, ch = k & 15;
            vx = w[co * 288 + ch * 9 + tap];          // ci = ch
            vy = w[co * 288 + (16 + ch) * 9 + tap];   // ci = 16 + ch
        }
        Wx[idx] = (_Float16)vx;
        Wy[idx] = (_Float16)vy;
    }

    // ---- zero the constant K-pad band [144,160) once ----
    for (int idx = tid; idx < DW * 16; idx += 256) {
        const int p = idx >> 4;
        patch[p * KSTR + KREAL + (idx & 15)] = (_Float16)0.0f;
    }

    // ---- x patch -> ox tile (+bias), held in registers ----
    build_patch(x + (long)b * DCIN * DH * DW, h, bpix, c2, patch);
    __syncthreads();

    v8f oxacc[4];
    #pragma unroll
    for (int ct = 0; ct < 4; ++ct) oxacc[ct] = (v8f){0,0,0,0,0,0,0,0};
    conv_tile(Wx, patch, aBase, bBase, oxacc);

    #pragma unroll
    for (int ct = 0; ct < 4; ++ct)
        #pragma unroll
        for (int i = 0; i < 8; ++i)
            oxacc[ct][i] += bias[ct * 16 + i + hi * 8];   // C/D: VGPR i -> M=i (+8 hi)

    v8f meanacc[4];
    #pragma unroll
    for (int ct = 0; ct < 4; ++ct) meanacc[ct] = (v8f){0,0,0,0,0,0,0,0};

    const long imgStride  = (long)DCIN * DH * DW;          // per (b,sy) image
    const long coStride   = (long)DH * DW;                 // 16384
    const long interBase  = (long)DB * DCO * DH * DW;      // new_target elems
    // store base for (b, sy=0, co = hi*8, h, pix)
    float* outp0 = out + interBase
                 + (((long)b * DSY * DCO + hi * 8) * DH + h) * DW + pix;
    const float* ybase = y + (long)b * DSY * imgStride;

    for (int sy = 0; sy < DSY; ++sy) {
        __syncthreads();                                   // previous patch consumed
        build_patch(ybase + (long)sy * imgStride, h, bpix, c2, patch);
        __syncthreads();

        if (sy + 1 < DSY) {                                // warm L2 for next image
            const float* nxt = ybase + (long)(sy + 1) * imgStride
                             + (long)(tid & 15) * coStride + (long)h * DW;
            __builtin_prefetch(nxt, 0, 1);
        }

        v8f acc[4];
        #pragma unroll
        for (int ct = 0; ct < 4; ++ct) acc[ct] = oxacc[ct];   // seed -> oz directly
        conv_tile(Wy, patch, aBase, bBase, acc);

        float* outp = outp0 + (long)sy * DCO * coStride;
        #pragma unroll
        for (int ct = 0; ct < 4; ++ct)
            #pragma unroll
            for (int i = 0; i < 8; ++i) {
                const float z = acc[ct][i];
                const float r = (z >= 0.0f) ? z : NEG_SLOPE * z;
                outp[(long)(ct * 16 + i) * coStride] = r;     // const idx -> IOFFSET
                meanacc[ct][i] += r;
            }
    }

    // ---- new_target = mean over sy ----
    float* mout = out + (((long)b * DCO + hi * 8) * DH + h) * DW + pix;
    #pragma unroll
    for (int ct = 0; ct < 4; ++ct)
        #pragma unroll
        for (int i = 0; i < 8; ++i)
            mout[(long)(ct * 16 + i) * coStride] = meanacc[ct][i] * (1.0f / (float)DSY);
}

extern "C" void kernel_launch(void* const* d_in, const int* in_sizes, int n_in,
                              void* d_out, int out_size, void* d_ws, size_t ws_size,
                              hipStream_t stream) {
    const float* x    = (const float*)d_in[0];   // (4,1,16,128,128)
    const float* y    = (const float*)d_in[1];   // (4,32,16,128,128)
    const float* wgt  = (const float*)d_in[2];   // (64,32,3,3)
    const float* bias = (const float*)d_in[3];   // (64,)
    float* out = (float*)d_out;

    dim3 grid(DH, DB);     // one block per (row h, batch b)
    dim3 block(256);       // 8 waves
    cross_conv_kernel<<<grid, block, SMEM_BYTES, stream>>>(x, y, wgt, bias, out);
}
